// M2M_8323646619754
// MI455X (gfx1250) — matrix operand order repaired and checked by reading, not compiled, not run on hardware
//
#include <hip/hip_runtime.h>
#include <hip/hip_bf16.h>

// ---------------------------------------------------------------------------
// Problem constants (from the reference)
// ---------------------------------------------------------------------------
constexpr int   CH        = 128;     // feature dim
constexpr int   NNODES    = 100000;  // nodes
constexpr int   NSCALES   = 6;
constexpr int   NLAYERS   = 4;
constexpr int   NEDGES    = 200000;  // edges per scale per direction

typedef __attribute__((ext_vector_type(16))) __bf16       v16bf;
typedef __attribute__((ext_vector_type(8)))  float        v8f;
typedef __attribute__((ext_vector_type(4)))  __bf16       bf16x4;
typedef __attribute__((ext_vector_type(4)))  unsigned int u32x4;

union AFrag { v16bf v; u32x4 u[2]; };

// ---------------------------------------------------------------------------
// GEMM: out[M,128] = A[M,128](bf16) @ W[128,128](bf16, row-major).T  (fp32 out)
// out[m,n] = sum_k A[m,k] * W[n,k]  -> B-matrix column n == W row n.
//
// Block = 256 threads (8 waves). W (32 KB bf16) is staged once into LDS and
// shared by all 8 waves; each wave computes a 16-row x 128-col strip with
// 32 v_wmma_f32_16x16x32_bf16. B-fragments are double-buffered in registers:
// the next N-tile's ds_loads are in flight while the current tile's 4-WMMA
// accumulation chain runs, enabling partial DScnt waits instead of drains.
// ---------------------------------------------------------------------------
__global__ __launch_bounds__(256) void gemm_bf16_n128(
    const __bf16* __restrict__ A, const __bf16* __restrict__ W,
    float* __restrict__ out, int M)
{
  __shared__ __bf16 wlds[CH * CH];          // 32 KB of 320 KB/WGP

  // Cooperative stage of W into LDS: 256 threads x 8 x b128.
  {
    const u32x4* src = (const u32x4*)W;
    u32x4*       dst = (u32x4*)wlds;
    int t = threadIdx.x;
#pragma unroll
    for (int i = 0; i < 8; ++i)
      dst[t + i * 256] = src[t + i * 256];
  }
  __syncthreads();

  const int wave = blockIdx.x * (blockDim.x >> 5) + (threadIdx.x >> 5);
  const int row0 = wave << 4;
  if (row0 >= M) return;                 // whole-wave exit: EXEC stays all-1s
  const int lane = threadIdx.x & 31;
  const int r    = lane & 15;
  const int hi   = lane >> 4;

  // A fragments for all 4 K-chunks (ISA 16-bit A 16x32 layout):
  // lane<16: K = kc*32 + {0..7, 16..23}; lane>=16: +8.
  AFrag a[4];
  const __bf16* arow = A + (size_t)(row0 + r) * CH + 8 * hi;
#pragma unroll
  for (int kc = 0; kc < 4; ++kc) {
    a[kc].u[0] = *(const u32x4*)(arow + kc * 32);
    a[kc].u[1] = *(const u32x4*)(arow + kc * 32 + 16);
  }

  // B fragment base (ISA 16-bit B 32x16 layout): lane = column n,
  // K = {0..7, 16..23} (+8 for lane-half 1) of W row (nt*16 + r).
  const __bf16* wbase = wlds + (size_t)r * CH + 8 * hi;

  AFrag b_cur[4], b_nxt[4];
#pragma unroll
  for (int kc = 0; kc < 4; ++kc) {         // prologue: load N-tile 0
    b_cur[kc].u[0] = *(const u32x4*)(wbase + kc * 32);
    b_cur[kc].u[1] = *(const u32x4*)(wbase + kc * 32 + 16);
  }

#pragma unroll
  for (int nt = 0; nt < 8; ++nt) {
    // Prefetch next N-tile's B fragments while this tile computes.
    if (nt + 1 < 8) {
      const __bf16* wrow = wbase + (size_t)(nt + 1) * 16 * CH;
#pragma unroll
      for (int kc = 0; kc < 4; ++kc) {
        b_nxt[kc].u[0] = *(const u32x4*)(wrow + kc * 32);
        b_nxt[kc].u[1] = *(const u32x4*)(wrow + kc * 32 + 16);
      }
    }

    v8f acc = {};
#pragma unroll
    for (int kc = 0; kc < 4; ++kc)         // back-to-back accumulation chain
      acc = __builtin_amdgcn_wmma_f32_16x16x32_bf16(
          false, a[kc].v, false, b_cur[kc].v, (short)0, acc, false, false);

    // C/D layout: VGPR j -> row m = row0 + j + 8*hi, col n = nt*16 + r
    float* orow = out + (size_t)row0 * CH + nt * 16 + r;
#pragma unroll
    for (int j = 0; j < 8; ++j)
      orow[(size_t)(j + 8 * hi) * CH] = acc[j];

#pragma unroll
    for (int kc = 0; kc < 4; ++kc)         // rotate (register rename)
      b_cur[kc] = b_nxt[kc];
  }
}

// ---------------------------------------------------------------------------
// fp32 -> bf16 conversion, 4 elements / thread
// ---------------------------------------------------------------------------
__global__ __launch_bounds__(256) void cvt_f32_bf16x4(
    const float* __restrict__ in, __bf16* __restrict__ out, long n4)
{
  long i = (long)blockIdx.x * blockDim.x + threadIdx.x;
  if (i >= n4) return;
  float4 f = ((const float4*)in)[i];
  bf16x4 o;
  o[0] = (__bf16)f.x; o[1] = (__bf16)f.y; o[2] = (__bf16)f.z; o[3] = (__bf16)f.w;
  ((bf16x4*)out)[i] = o;
}

// ---------------------------------------------------------------------------
// Edge scatter-add: temp[U[e], :] += h[V[e], :]. One wave per edge,
// float4 per lane, global_atomic_add_f32 into L2-resident temp.
// ---------------------------------------------------------------------------
__global__ __launch_bounds__(256) void scatter_add_rows(
    const int* __restrict__ U, const int* __restrict__ V,
    const float* __restrict__ h, float* __restrict__ temp, int E)
{
  int t = blockIdx.x * blockDim.x + threadIdx.x;
  int e = t >> 5;
  if (e >= E) return;
  int c   = (t & 31) << 2;
  int src = V[e];
  int dst = U[e];
  float4 val = *(const float4*)(h + (size_t)src * CH + c);
  float* d   = temp + (size_t)dst * CH + c;
  atomicAdd(d + 0, val.x);
  atomicAdd(d + 1, val.y);
  atomicAdd(d + 2, val.z);
  atomicAdd(d + 3, val.w);
}

// ---------------------------------------------------------------------------
// Wave (32-lane) sum reduction
// ---------------------------------------------------------------------------
__device__ __forceinline__ float wave_sum(float s) {
#pragma unroll
  for (int off = 16; off > 0; off >>= 1) s += __shfl_xor(s, off, 32);
  return s;
}

// GroupNorm(1 group over 128ch) + ReLU, output bf16 (feeds next GEMM).
// One wave per row, 4 channels per lane.
__global__ __launch_bounds__(256) void gn_relu_bf16(
    const float* __restrict__ x, const float* __restrict__ g,
    const float* __restrict__ b, __bf16* __restrict__ out, int Nrows)
{
  int row = blockIdx.x * 8 + (threadIdx.x >> 5);
  if (row >= Nrows) return;
  int lane = threadIdx.x & 31;
  int c = lane << 2;
  float4 v = *(const float4*)(x + (size_t)row * CH + c);
  float mu = wave_sum(v.x + v.y + v.z + v.w) * (1.0f / CH);
  float d0 = v.x - mu, d1 = v.y - mu, d2 = v.z - mu, d3 = v.w - mu;
  float var = wave_sum(d0 * d0 + d1 * d1 + d2 * d2 + d3 * d3) * (1.0f / CH);
  float inv = rsqrtf(var + 1e-5f);
  float4 gg = *(const float4*)(g + c);
  float4 bb = *(const float4*)(b + c);
  bf16x4 o;
  o[0] = (__bf16)fmaxf(d0 * inv * gg.x + bb.x, 0.0f);
  o[1] = (__bf16)fmaxf(d1 * inv * gg.y + bb.y, 0.0f);
  o[2] = (__bf16)fmaxf(d2 * inv * gg.z + bb.z, 0.0f);
  o[3] = (__bf16)fmaxf(d3 * inv * gg.w + bb.w, 0.0f);
  *(bf16x4*)(out + (size_t)row * CH + c) = o;
}

// GroupNorm(y) + residual + ReLU -> new feat (fp32) and bf16 copy for the
// next layer's GEMMs.
__global__ __launch_bounds__(256) void gn2_res_relu(
    const float* __restrict__ y, const float* __restrict__ res,
    const float* __restrict__ g, const float* __restrict__ b,
    float* __restrict__ feat_out, __bf16* __restrict__ featb_out, int Nrows)
{
  int row = blockIdx.x * 8 + (threadIdx.x >> 5);
  if (row >= Nrows) return;
  int lane = threadIdx.x & 31;
  int c = lane << 2;
  float4 v = *(const float4*)(y + (size_t)row * CH + c);
  float mu = wave_sum(v.x + v.y + v.z + v.w) * (1.0f / CH);
  float d0 = v.x - mu, d1 = v.y - mu, d2 = v.z - mu, d3 = v.w - mu;
  float var = wave_sum(d0 * d0 + d1 * d1 + d2 * d2 + d3 * d3) * (1.0f / CH);
  float inv = rsqrtf(var + 1e-5f);
  float4 gg = *(const float4*)(g + c);
  float4 bb = *(const float4*)(b + c);
  float4 rr = *(const float4*)(res + (size_t)row * CH + c);
  float4 o;
  o.x = fmaxf(d0 * inv * gg.x + bb.x + rr.x, 0.0f);
  o.y = fmaxf(d1 * inv * gg.y + bb.y + rr.y, 0.0f);
  o.z = fmaxf(d2 * inv * gg.z + bb.z + rr.z, 0.0f);
  o.w = fmaxf(d3 * inv * gg.w + bb.w + rr.w, 0.0f);
  *(float4*)(feat_out + (size_t)row * CH + c) = o;
  bf16x4 ob;
  ob[0] = (__bf16)o.x; ob[1] = (__bf16)o.y; ob[2] = (__bf16)o.z; ob[3] = (__bf16)o.w;
  *(bf16x4*)(featb_out + (size_t)row * CH + c) = ob;
}

// ---------------------------------------------------------------------------
// Host orchestration
// ---------------------------------------------------------------------------
extern "C" void kernel_launch(void* const* d_in, const int* in_sizes, int n_in,
                              void* d_out, int out_size, void* d_ws, size_t ws_size,
                              hipStream_t stream)
{
  (void)in_sizes; (void)n_in; (void)out_size; (void)ws_size;

  const float* feat_in = (const float*)d_in[0];
  const float* W_ctr   = (const float*)d_in[1];   // [4,128,128]
  const float* W_pre   = (const float*)d_in[2];   // [6,4,128,128]
  const float* W_suc   = (const float*)d_in[3];   // [6,4,128,128]
  const float* norm_g  = (const float*)d_in[4];   // [4,128]
  const float* norm_b  = (const float*)d_in[5];
  const float* W_ctr2  = (const float*)d_in[6];   // [4,128,128]
  const float* ctr2_g  = (const float*)d_in[7];
  const float* ctr2_b  = (const float*)d_in[8];
  const int*   pre_u   = (const int*)d_in[9];     // [6,E]
  const int*   pre_v   = (const int*)d_in[10];
  const int*   suc_u   = (const int*)d_in[11];
  const int*   suc_v   = (const int*)d_in[12];

  // Workspace carve-out
  char* ws = (char*)d_ws;
  size_t off = 0;
  auto carve = [&](size_t bytes) -> void* {
    void* p = ws + off;
    off = (off + bytes + 255) & ~(size_t)255;
    return p;
  };
  const size_t NC  = (size_t)NNODES * CH;
  const size_t WSZ = (size_t)CH * CH;
  float*  featbuf = (float*)  carve(NC * 4);
  __bf16* featb   = (__bf16*) carve(NC * 2);
  float*  temp    = (float*)  carve(NC * 4);
  float*  hbuf    = (float*)  carve(NC * 4);
  __bf16* wctr_b  = (__bf16*) carve((size_t)NLAYERS * WSZ * 2);
  __bf16* wpre_b  = (__bf16*) carve((size_t)NSCALES * NLAYERS * WSZ * 2);
  __bf16* wsuc_b  = (__bf16*) carve((size_t)NSCALES * NLAYERS * WSZ * 2);
  __bf16* wctr2_b = (__bf16*) carve((size_t)NLAYERS * WSZ * 2);
  (void)featbuf;

  auto cvt = [&](const float* src, __bf16* dst, size_t n) {
    long n4 = (long)(n >> 2);
    cvt_f32_bf16x4<<<(int)((n4 + 255) / 256), 256, 0, stream>>>(src, dst, n4);
  };

  // One-time bf16 conversions (re-done every call: deterministic)
  cvt(W_ctr,  wctr_b,  (size_t)NLAYERS * WSZ);
  cvt(W_pre,  wpre_b,  (size_t)NSCALES * NLAYERS * WSZ);
  cvt(W_suc,  wsuc_b,  (size_t)NSCALES * NLAYERS * WSZ);
  cvt(W_ctr2, wctr2_b, (size_t)NLAYERS * WSZ);
  cvt(feat_in, featb, NC);

  const int gemm_blocks = (NNODES / 16 + 7) / 8;            // 8 waves/block
  const int scat_blocks = (NEDGES * 32 + 255) / 256;        // wave per edge
  const int gn_blocks   = (NNODES + 7) / 8;                 // wave per row

  for (int L = 0; L < NLAYERS; ++L) {
    // temp = feat @ W_ctr[L].T
    gemm_bf16_n128<<<gemm_blocks, 256, 0, stream>>>(
        featb, wctr_b + (size_t)L * WSZ, temp, NNODES);

    // Multi-scale aggregation: transform ALL nodes once, then gather/scatter.
    for (int s = 0; s < NSCALES; ++s) {
      gemm_bf16_n128<<<gemm_blocks, 256, 0, stream>>>(
          featb, wpre_b + (size_t)(s * NLAYERS + L) * WSZ, hbuf, NNODES);
      scatter_add_rows<<<scat_blocks, 256, 0, stream>>>(
          pre_u + (size_t)s * NEDGES, pre_v + (size_t)s * NEDGES,
          hbuf, temp, NEDGES);
      gemm_bf16_n128<<<gemm_blocks, 256, 0, stream>>>(
          featb, wsuc_b + (size_t)(s * NLAYERS + L) * WSZ, hbuf, NNODES);
      scatter_add_rows<<<scat_blocks, 256, 0, stream>>>(
          suc_u + (size_t)s * NEDGES, suc_v + (size_t)s * NEDGES,
          hbuf, temp, NEDGES);
    }

    // x = relu(GN(temp))  -> bf16 (overwrites featb; no longer needed)
    gn_relu_bf16<<<gn_blocks, 256, 0, stream>>>(
        temp, norm_g + (size_t)L * CH, norm_b + (size_t)L * CH, featb, NNODES);

    // y = x @ W_ctr2[L].T
    gemm_bf16_n128<<<gemm_blocks, 256, 0, stream>>>(
        featb, wctr2_b + (size_t)L * WSZ, hbuf, NNODES);

    // feat = relu(GN(y) + res); res = feat
    const float* resp = (L == 0) ? feat_in : featbuf;
    float* fout = (L == NLAYERS - 1) ? (float*)d_out : featbuf;
    gn2_res_relu<<<gn_blocks, 256, 0, stream>>>(
        hbuf, resp, ctr2_g + (size_t)L * CH, ctr2_b + (size_t)L * CH,
        fout, featb, NNODES);
  }
}
